// RoPEAttentionHead_40364102648143
// MI455X (gfx1250) — matrix operand order repaired and checked by reading, not compiled, not run on hardware
//
#include <hip/hip_runtime.h>
#include <hip/hip_bf16.h>

// ---------------------------------------------------------------------------
// RoPE attention head, B=8, M=2048, D=1024, fp32 in/out.
// bf16 WMMA (v_wmma_f32_16x16x32_bf16) for all GEMMs, flash-style attention.
// ---------------------------------------------------------------------------

typedef __attribute__((ext_vector_type(16))) __bf16 bf16x16;
typedef __attribute__((ext_vector_type(8)))  __bf16 bf16x8;
typedef __attribute__((ext_vector_type(4)))  __bf16 bf16x4;
typedef __attribute__((ext_vector_type(8)))  float  f32x8;

union FragAB { bf16x16 v; bf16x8 h[2]; };

constexpr int   Bn   = 8;
constexpr int   Mseq = 2048;
constexpr int   Dd   = 1024;
constexpr int   MR   = Bn * Mseq;          // 16384 total rows

// ---------------------------------------------------------------------------
// Kernel 1: convert x (16M f32) and wq/wk/wv (1M f32 each) to bf16.
// ---------------------------------------------------------------------------
__global__ __launch_bounds__(256)
void cvt_kernel(const float* __restrict__ x,
                const float* __restrict__ wq,
                const float* __restrict__ wk,
                const float* __restrict__ wv,
                __bf16* __restrict__ xb,
                __bf16* __restrict__ wqb,
                __bf16* __restrict__ wkb,
                __bf16* __restrict__ wvb) {
    const size_t NX = (size_t)MR * Dd;       // 16,777,216
    const size_t NW = (size_t)Dd * Dd;       //  1,048,576
    size_t i = ((size_t)blockIdx.x * blockDim.x + threadIdx.x) * 4;
    if (i < NX) {
        float4 v = *(const float4*)(x + i);
        bf16x4 o; o[0] = (__bf16)v.x; o[1] = (__bf16)v.y; o[2] = (__bf16)v.z; o[3] = (__bf16)v.w;
        *(bf16x4*)(xb + i) = o;
    }
    if (i < NW) {
        float4 a = *(const float4*)(wq + i);
        bf16x4 oa; oa[0] = (__bf16)a.x; oa[1] = (__bf16)a.y; oa[2] = (__bf16)a.z; oa[3] = (__bf16)a.w;
        *(bf16x4*)(wqb + i) = oa;
        float4 b = *(const float4*)(wk + i);
        bf16x4 ob; ob[0] = (__bf16)b.x; ob[1] = (__bf16)b.y; ob[2] = (__bf16)b.z; ob[3] = (__bf16)b.w;
        *(bf16x4*)(wkb + i) = ob;
        float4 c = *(const float4*)(wv + i);
        bf16x4 oc; oc[0] = (__bf16)c.x; oc[1] = (__bf16)c.y; oc[2] = (__bf16)c.z; oc[3] = (__bf16)c.w;
        *(bf16x4*)(wvb + i) = oc;
    }
}

// ---------------------------------------------------------------------------
// Shared epilogue for one 16x16 C tile of the projection GEMM.
//   which < 2 : apply RoPE (pair exchange via shfl_xor lane^1) and store bf16.
//   which == 2: store V transposed per batch: vT[b][d][m].
// ---------------------------------------------------------------------------
__device__ __forceinline__
void proj_epilogue_tile(const f32x8& c, int mt0, int nt0, int which,
                        __bf16* __restrict__ qk, __bf16* __restrict__ vT,
                        int hl, int l15) {
    if (which < 2) {
        const int col = nt0 + l15;
        const float fi    = (float)(col >> 1);
        const float theta = __powf(10000.0f, -2.0f * (fi - 1.0f) * (1.0f / (float)Dd));
        #pragma unroll
        for (int r = 0; r < 8; ++r) {
            const int grow = mt0 + r + hl * 8;
            const int mpos = grow & (Mseq - 1);
            const float val     = c[r];
            const float partner = __shfl_xor(val, 1, 32);
            float sv, cv;
            __sincosf((float)mpos * theta, &sv, &cv);
            // even col: re =  te*cos + to*sin   (val=te, partner=to)
            // odd  col: ro = -te*sin + to*cos   (val=to, partner=te)
            const float res = ((col & 1) == 0) ? (val * cv + partner * sv)
                                               : (val * cv - partner * sv);
            qk[(size_t)grow * Dd + col] = (__bf16)res;
        }
    } else {
        #pragma unroll
        for (int r = 0; r < 8; ++r) {
            const int grow = mt0 + r + hl * 8;
            const int bat  = grow >> 11;
            const int mpos = grow & (Mseq - 1);
            vT[((size_t)bat * Dd + (nt0 + l15)) * Mseq + mpos] = (__bf16)c[r];
        }
    }
}

// ---------------------------------------------------------------------------
// Kernel 2: projections q/k/v = x @ w^T.  One wave per 32x32 output block
// (2x2 register tiling: 2 A-frags + 2 B-frags feed 4 WMMAs per K-step, i.e.
// 2 b128 loads per WMMA instead of 4).
//   which = blockIdx.z: 0->q (RoPE), 1->k (RoPE), 2->v (stored transposed).
//   A-frag: lane holds row (lane&15), K = hl*8 + {0..7, 16..23}  (two b128)
//   B-frag: lane holds col (lane&15) of B = w row; K = hl*8 + {0..7, 16..23} (two b128)
// ---------------------------------------------------------------------------
__global__ __launch_bounds__(256)
void proj_kernel(const __bf16* __restrict__ xb,
                 const __bf16* __restrict__ wqb,
                 const __bf16* __restrict__ wkb,
                 const __bf16* __restrict__ wvb,
                 __bf16* __restrict__ qr,
                 __bf16* __restrict__ kr,
                 __bf16* __restrict__ vT) {
    const int lane = threadIdx.x & 31;
    const int wv32 = threadIdx.x >> 5;
    const int hl   = lane >> 4;
    const int l15  = lane & 15;
    const int which = blockIdx.z;
    const int m0 = blockIdx.y * 32;                    // 32-row block
    const int n0 = (blockIdx.x * 8 + wv32) * 32;       // 32-col block per wave
    const __bf16* w = (which == 0) ? wqb : (which == 1) ? wkb : wvb;

    const __bf16* arow0 = xb + (size_t)(m0 + l15) * Dd;
    const __bf16* arow1 = arow0 + (size_t)16 * Dd;
    const __bf16* brow0 = w  + (size_t)(n0 + l15) * Dd;
    const __bf16* brow1 = brow0 + (size_t)16 * Dd;

    f32x8 c00 = {}, c01 = {}, c10 = {}, c11 = {};
    for (int k0 = 0; k0 < Dd; k0 += 32) {
        FragAB a0, a1, b0, b1;
        a0.h[0] = *(const bf16x8*)(arow0 + k0 + hl * 8);
        a0.h[1] = *(const bf16x8*)(arow0 + k0 + hl * 8 + 16);
        a1.h[0] = *(const bf16x8*)(arow1 + k0 + hl * 8);
        a1.h[1] = *(const bf16x8*)(arow1 + k0 + hl * 8 + 16);
        b0.h[0] = *(const bf16x8*)(brow0 + k0 + hl * 8);
        b0.h[1] = *(const bf16x8*)(brow0 + k0 + hl * 8 + 16);
        b1.h[0] = *(const bf16x8*)(brow1 + k0 + hl * 8);
        b1.h[1] = *(const bf16x8*)(brow1 + k0 + hl * 8 + 16);
        c00 = __builtin_amdgcn_wmma_f32_16x16x32_bf16(false, a0.v, false, b0.v, (short)0, c00, false, false);
        c01 = __builtin_amdgcn_wmma_f32_16x16x32_bf16(false, a0.v, false, b1.v, (short)0, c01, false, false);
        c10 = __builtin_amdgcn_wmma_f32_16x16x32_bf16(false, a1.v, false, b0.v, (short)0, c10, false, false);
        c11 = __builtin_amdgcn_wmma_f32_16x16x32_bf16(false, a1.v, false, b1.v, (short)0, c11, false, false);
    }

    __bf16* qk = (which == 0) ? qr : kr;
    proj_epilogue_tile(c00, m0,      n0,      which, qk, vT, hl, l15);
    proj_epilogue_tile(c01, m0,      n0 + 16, which, qk, vT, hl, l15);
    proj_epilogue_tile(c10, m0 + 16, n0,      which, qk, vT, hl, l15);
    proj_epilogue_tile(c11, m0 + 16, n0 + 16, which, qk, vT, hl, l15);
}

// ---------------------------------------------------------------------------
// Kernel 3: flash attention. One workgroup (8 waves) per (batch, 16-row qtile).
// Wave w: computes S columns [w*16, w*16+16) of each 128-key block, and owns
// output columns [w*128, (w+1)*128) (8 accumulator tiles).
// ---------------------------------------------------------------------------
__global__ __launch_bounds__(256)
void attn_kernel(const __bf16* __restrict__ qr,
                 const __bf16* __restrict__ kr,
                 const __bf16* __restrict__ vT,
                 float* __restrict__ out) {
    __shared__ __align__(16) __bf16 qs[16 * Dd];     // 32 KB: q tile (rotated)
    __shared__ __align__(16) __bf16 ps[16 * 128];    //  4 KB: P block (bf16)
    __shared__ float mrow[16], lrow[16], fac[16];
    __shared__ float smax[8][16], ssum[8][16];

    const int tid  = threadIdx.x;
    const int lane = tid & 31;
    const int wv32 = tid >> 5;
    const int hl   = lane >> 4;
    const int l15  = lane & 15;
    const int bat  = blockIdx.x >> 7;                // 128 q tiles per batch
    const int mq0  = (blockIdx.x & 127) * 16;
    const size_t rowbase = (size_t)bat * Mseq + mq0;

    // Stage rotated-Q tile (16 x 1024 bf16) into LDS.
    for (int i = tid; i < 16 * Dd / 8; i += 256) {
        const int r  = i >> 7;
        const int cc = (i & 127) * 8;
        *(bf16x8*)&qs[r * Dd + cc] = *(const bf16x8*)&qr[(rowbase + r) * Dd + cc];
    }
    if (tid < 16) { mrow[tid] = -3.0e38f; lrow[tid] = 0.0f; }

    f32x8 acc[8];
    #pragma unroll
    for (int t = 0; t < 8; ++t) acc[t] = (f32x8){};

    const int d0base = wv32 * 128;
    const __bf16* vbase = vT + (size_t)bat * Dd * Mseq;

    for (int kb = 0; kb < Mseq / 128; ++kb) {
        __syncthreads();   // protects ps/smax reuse, and first-iter qs readiness

        // ---- scores S (16 x 16 per wave): rows = q rows, cols = keys ----
        const int j0 = kb * 128 + wv32 * 16;
        const __bf16* krow = kr + ((size_t)bat * Mseq + j0 + l15) * Dd;
        f32x8 s = {};
        for (int k0 = 0; k0 < Dd; k0 += 32) {
            FragAB a, b;
            a.h[0] = *(const bf16x8*)&qs[l15 * Dd + k0 + hl * 8];
            a.h[1] = *(const bf16x8*)&qs[l15 * Dd + k0 + hl * 8 + 16];
            b.h[0] = *(const bf16x8*)(krow + k0 + hl * 8);
            b.h[1] = *(const bf16x8*)(krow + k0 + hl * 8 + 16);
            s = __builtin_amdgcn_wmma_f32_16x16x32_bf16(false, a.v, false, b.v,
                                                        (short)0, s, false, false);
        }
        // scale + per-row max over this wave's 16 columns
        #pragma unroll
        for (int r = 0; r < 8; ++r) {
            float v = s[r] * 0.03125f;     // 1/sqrt(1024)
            s[r] = v;
            #pragma unroll
            for (int off = 1; off < 16; off <<= 1) v = fmaxf(v, __shfl_xor(v, off, 32));
            if (l15 == 0) smax[wv32][r + hl * 8] = v;
        }
        __syncthreads();

        // combine running max, compute rescale factor
        if (tid < 16) {
            const float mo = mrow[tid];
            float mn = mo;
            #pragma unroll
            for (int w2 = 0; w2 < 8; ++w2) mn = fmaxf(mn, smax[w2][tid]);
            const float f = __expf(mo - mn);
            fac[tid]  = f;
            mrow[tid] = mn;
            lrow[tid] *= f;
        }
        __syncthreads();

        // rescale acc, P = exp(s - m), row sums, write P (bf16) to LDS
        #pragma unroll
        for (int r = 0; r < 8; ++r) {
            const int row = r + hl * 8;
            const float f = fac[row];
            #pragma unroll
            for (int t = 0; t < 8; ++t) acc[t][r] *= f;
            const float p = __expf(s[r] - mrow[row]);
            float v = p;
            #pragma unroll
            for (int off = 1; off < 16; off <<= 1) v += __shfl_xor(v, off, 32);
            if (l15 == 0) ssum[wv32][row] = v;
            ps[row * 128 + wv32 * 16 + l15] = (__bf16)p;
        }
        __syncthreads();

        if (tid < 16) {
            float l = lrow[tid];
            #pragma unroll
            for (int w2 = 0; w2 < 8; ++w2) l += ssum[w2][tid];
            lrow[tid] = l;
        }

        // ---- O update: acc[t] += P(16x128) @ V(128x16), A-frags from LDS ----
        #pragma unroll
        for (int kk = 0; kk < 4; ++kk) {
            FragAB a;
            a.h[0] = *(const bf16x8*)&ps[l15 * 128 + kk * 32 + hl * 8];
            a.h[1] = *(const bf16x8*)&ps[l15 * 128 + kk * 32 + hl * 8 + 16];
            #pragma unroll
            for (int t = 0; t < 8; ++t) {
                const int d = d0base + t * 16 + l15;
                const __bf16* vcol = vbase + (size_t)d * Mseq + kb * 128 + kk * 32;
                FragAB b;
                b.h[0] = *(const bf16x8*)(vcol + hl * 8);
                b.h[1] = *(const bf16x8*)(vcol + hl * 8 + 16);
                acc[t] = __builtin_amdgcn_wmma_f32_16x16x32_bf16(false, a.v, false, b.v,
                                                                 (short)0, acc[t], false, false);
            }
        }
    }
    __syncthreads();

    // epilogue: divide by softmax denominator, store f32
    #pragma unroll
    for (int r = 0; r < 8; ++r) {
        const int row = r + hl * 8;
        const float inv = 1.0f / lrow[row];
        const size_t obase = (rowbase + row) * Dd;
        #pragma unroll
        for (int t = 0; t < 8; ++t)
            out[obase + d0base + t * 16 + l15] = acc[t][r] * inv;
    }
}

// ---------------------------------------------------------------------------
// Host launcher. Workspace layout (bytes), total ~134 MB:
//   [  0,  32M) xb      bf16 x
//   [ 32M, 34M) wqb     [34M,36M) wkb   [36M,38M) wvb
//   [ 38M, 70M) qr      rotated Q, bf16, [B*M][D]
//   [ 70M,102M) kr      rotated K, bf16, [B*M][D]
//   [102M,134M) vT      V transposed, bf16, [B][D][M]
// ---------------------------------------------------------------------------
extern "C" void kernel_launch(void* const* d_in, const int* in_sizes, int n_in,
                              void* d_out, int out_size, void* d_ws, size_t ws_size,
                              hipStream_t stream) {
    const float* x  = (const float*)d_in[0];
    const float* wq = (const float*)d_in[1];
    const float* wk = (const float*)d_in[2];
    const float* wv = (const float*)d_in[3];
    float* out = (float*)d_out;

    char* ws = (char*)d_ws;
    const size_t MB = (size_t)1 << 20;
    __bf16* xb  = (__bf16*)(ws);
    __bf16* wqb = (__bf16*)(ws + 32 * MB);
    __bf16* wkb = (__bf16*)(ws + 34 * MB);
    __bf16* wvb = (__bf16*)(ws + 36 * MB);
    __bf16* qr  = (__bf16*)(ws + 38 * MB);
    __bf16* kr  = (__bf16*)(ws + 70 * MB);
    __bf16* vT  = (__bf16*)(ws + 102 * MB);

    // 1) f32 -> bf16 conversion (16M + 3x1M elements, 4 per thread)
    cvt_kernel<<<dim3(16384), dim3(256), 0, stream>>>(x, wq, wk, wv, xb, wqb, wkb, wvb);

    // 2) Q/K/V projections + fused RoPE / transpose.
    //    Each wave: 32x32 output block; block of 8 waves covers 256 cols.
    //    grid: x = 1024/256 = 4, y = 16384/32 = 512 row blocks, z = 3 (q,k,v)
    proj_kernel<<<dim3(4, MR / 32, 3), dim3(256), 0, stream>>>(xb, wqb, wkb, wvb, qr, kr, vT);

    // 3) flash attention: one block per (batch, 16-row q tile)
    attn_kernel<<<dim3(Bn * Mseq / 16), dim3(256), 0, stream>>>(qr, kr, vT, out);
}
